// ZINB_CSVAE_32658931319273
// MI455X (gfx1250) — compile-verified, matrix operand order checked by reading, not run on hardware
//
#include <hip/hip_runtime.h>
#include <math.h>

typedef __bf16 bf16_t;
typedef __attribute__((ext_vector_type(16))) __bf16 v16bf;
typedef __attribute__((ext_vector_type(8)))  __bf16 bf16x8;
typedef __attribute__((ext_vector_type(8)))  float  v8f;

#define NB   4096   // batch
#define NG   4000   // genes
#define NY   10
#define NZ   32
#define NW   80
#define NH   1024
#define NZW  112

__device__ __forceinline__ float softplusf(float x) {
    return fmaxf(x, 0.f) + log1pf(expf(-fabsf(x)));
}
__device__ __forceinline__ float logsigf(float x) {
    return fminf(x, 0.f) - log1pf(expf(-fabsf(x)));
}

// ---------------------------------------------------------------------------
// bf16 WMMA GEMM: C[M,N] = act(A[M,K] @ B[K,N] + bias)
// A : row-major bf16 [M, lda]  (K padded to mult of 32, pad cols zero)
// Bt: TRANSPOSED weights, row-major bf16 [Np, ldbt] = B^T (pad rows/cols zero)
// Block = 256 threads = 8 waves; tile 128(M) x 128(N) x 32(K);
// each wave owns a 32x64 subtile = 2x4 v_wmma_f32_16x16x32_bf16 accumulators.
// Double-buffered LDS: next K-tile's global loads issue before the WMMAs and
// retire underneath them; one barrier per K-step.
// ---------------------------------------------------------------------------
__global__ __launch_bounds__(256)
void gemm_bf16_kernel(const bf16_t* __restrict__ A, int lda,
                      const bf16_t* __restrict__ Bt, int ldbt,
                      const float* __restrict__ bias,
                      float* __restrict__ Cf, int ldc,
                      bf16_t* __restrict__ Cb, int ldcb,
                      int M, int N, int K, int act)
{
    __shared__ bf16_t As[2][128][32];   // 2 x 8 KB
    __shared__ bf16_t Bs[2][128][32];   // 2 x 8 KB, [n][k] (transposed in memory)

    const int tid  = threadIdx.x;
    const int lane = tid & 31;
    const int wave = tid >> 5;
    const int wm   = (wave & 3) * 32;   // 4 waves along M
    const int wn   = (wave >> 2) * 64;  // 2 waves along N
    const int bm   = blockIdx.y * 128;
    const int bn   = blockIdx.x * 128;

    const int sr = tid >> 1;            // staging row 0..127 (A row / B^T row)
    const int sc = (tid & 1) * 16;      // staging col base 0/16 (along K)

    const bf16_t* aRow = A  + (size_t)(bm + sr) * lda  + sc;
    const bf16_t* bRow = Bt + (size_t)(bn + sr) * ldbt + sc;

    // preload K-tile 0 -> registers -> LDS buffer 0
    bf16x8 ra0 = *(const bf16x8*)(aRow);
    bf16x8 ra1 = *(const bf16x8*)(aRow + 8);
    bf16x8 rb0 = *(const bf16x8*)(bRow);
    bf16x8 rb1 = *(const bf16x8*)(bRow + 8);
    *(bf16x8*)(&As[0][sr][sc])     = ra0;
    *(bf16x8*)(&As[0][sr][sc + 8]) = ra1;
    *(bf16x8*)(&Bs[0][sr][sc])     = rb0;
    *(bf16x8*)(&Bs[0][sr][sc + 8]) = rb1;
    __syncthreads();

    v8f acc[2][4] = {};
    int buf = 0;

    for (int k0 = 0; k0 < K; k0 += 32) {
        const bool hasNext = (k0 + 32) < K;
        if (hasNext) {
            // issue next tile's 4 global loads early; they complete under WMMAs
            ra0 = *(const bf16x8*)(aRow + k0 + 32);
            ra1 = *(const bf16x8*)(aRow + k0 + 40);
            rb0 = *(const bf16x8*)(bRow + k0 + 32);
            rb1 = *(const bf16x8*)(bRow + k0 + 40);
            if (k0 + 64 < K) {  // warm near caches for the tile after next
                __builtin_prefetch(aRow + k0 + 64, 0, 3);
                __builtin_prefetch(bRow + k0 + 64, 0, 3);
            }
        }

        // fragment gather per ISA 16-bit A/B layout:
        // lanes 0-15: K 0..7 (v0-3), 16..23 (v4-7); lanes 16-31: +8
        const int row = lane & 15;
        const int kh  = (lane >> 4) * 8;
        v16bf afrag[2], bfrag[4];
        #pragma unroll
        for (int mi = 0; mi < 2; ++mi) {
            const bf16_t* p = &As[buf][wm + mi * 16 + row][0];
            bf16x8 lo = *(const bf16x8*)(p + kh);
            bf16x8 hi = *(const bf16x8*)(p + 16 + kh);
            #pragma unroll
            for (int i = 0; i < 8; ++i) { afrag[mi][i] = lo[i]; afrag[mi][8 + i] = hi[i]; }
        }
        #pragma unroll
        for (int ni = 0; ni < 4; ++ni) {
            const bf16_t* p = &Bs[buf][wn + ni * 16 + row][0];
            bf16x8 lo = *(const bf16x8*)(p + kh);
            bf16x8 hi = *(const bf16x8*)(p + 16 + kh);
            #pragma unroll
            for (int i = 0; i < 8; ++i) { bfrag[ni][i] = lo[i]; bfrag[ni][8 + i] = hi[i]; }
        }

        #pragma unroll
        for (int mi = 0; mi < 2; ++mi)
            #pragma unroll
            for (int ni = 0; ni < 4; ++ni)
                acc[mi][ni] = __builtin_amdgcn_wmma_f32_16x16x32_bf16(
                    false, afrag[mi], false, bfrag[ni],
                    (short)0, acc[mi][ni], false, false);

        if (hasNext) {
            const int nb = buf ^ 1;
            *(bf16x8*)(&As[nb][sr][sc])     = ra0;
            *(bf16x8*)(&As[nb][sr][sc + 8]) = ra1;
            *(bf16x8*)(&Bs[nb][sr][sc])     = rb0;
            *(bf16x8*)(&Bs[nb][sr][sc + 8]) = rb1;
            __syncthreads();
            buf = nb;
        }
    }

    // epilogue: C/D layout — VGPR r: M = r + 8*(lane>=16), N = lane&15
    const int col  = lane & 15;
    const int rofs = (lane >> 4) * 8;
    #pragma unroll
    for (int mi = 0; mi < 2; ++mi) {
        #pragma unroll
        for (int ni = 0; ni < 4; ++ni) {
            #pragma unroll
            for (int r = 0; r < 8; ++r) {
                const int m = bm + wm + mi * 16 + rofs + r;
                const int n = bn + wn + ni * 16 + col;
                float v = acc[mi][ni][r];
                const bool in = (n < N) && (m < M);
                if (in) {
                    v += bias[n];
                    if (act == 1)      v = fmaxf(v, 0.f);
                    else if (act == 2) v = softplusf(v);
                    if (Cf) Cf[(size_t)m * ldc + n] = v;
                }
                if (Cb && m < M && n < ldcb)
                    Cb[(size_t)m * ldcb + n] = (bf16_t)(in ? v : 0.f);
            }
        }
    }
}

// ---------------------------------------------------------------------------
// Small dense layer, plain f32 VALU (used for the 32/80/112-wide MLPs).
// ---------------------------------------------------------------------------
__global__ void dense_f32_kernel(const float* __restrict__ X, int ldx,
                                 const float* __restrict__ Wm,
                                 const float* __restrict__ b,
                                 float* __restrict__ Yo, int ldy,
                                 int M, int N, int K, int act)
{
    int idx = blockIdx.x * 256 + threadIdx.x;
    if (idx >= M * N) return;
    int m = idx / N, n = idx % N;
    const float* xr = X + (size_t)m * ldx;
    float s = b[n];
    for (int k = 0; k < K; ++k) s += xr[k] * Wm[(size_t)k * N + n];
    if (act == 1)      s = fmaxf(s, 0.f);
    else if (act == 2) s = softplusf(s);
    Yo[(size_t)m * ldy + n] = s;
}

// f32 weight [K,N] -> zero-padded TRANSPOSED bf16 [Np,Kp]  (dst[n][k] = src[k][n])
__global__ void convert_wt_kernel(const float* __restrict__ src, bf16_t* __restrict__ dst,
                                  int K, int N, int Kp, int Np)
{
    int idx = blockIdx.x * 256 + threadIdx.x;
    if (idx >= Np * Kp) return;
    int n = idx / Kp, k = idx % Kp;
    float v = (k < K && n < N) ? src[(size_t)k * N + n] : 0.f;
    dst[idx] = (bf16_t)v;
}

// xy = [log1p(x) | y | 0-pad] in bf16, width 4032
__global__ void prep_xy_kernel(const float* __restrict__ x, const float* __restrict__ y,
                               bf16_t* __restrict__ xy)
{
    int idx = blockIdx.x * 256 + threadIdx.x;
    if (idx >= NB * 4032) return;
    int b = idx / 4032, c = idx % 4032;
    float v = 0.f;
    if (c < NG)            v = log1pf(x[(size_t)b * NG + c]);
    else if (c < NG + NY)  v = y[b * NY + (c - NG)];
    xy[idx] = (bf16_t)v;
}

// y -> bf16 padded to 32 cols
__global__ void prep_y_kernel(const float* __restrict__ y, bf16_t* __restrict__ yb)
{
    int idx = blockIdx.x * 256 + threadIdx.x;
    if (idx >= NB * 32) return;
    int b = idx >> 5, c = idx & 31;
    yb[idx] = (bf16_t)((c < NY) ? y[b * NY + c] : 0.f);
}

// f32 [M,N] (lds) -> bf16 [M,ldd] zero-padded
__global__ void cvt_act_kernel(const float* __restrict__ src, int lds,
                               bf16_t* __restrict__ dst, int ldd, int N, int total)
{
    int idx = blockIdx.x * 256 + threadIdx.x;
    if (idx >= total) return;
    int m = idx / ldd, n = idx % ldd;
    dst[idx] = (bf16_t)((n < N) ? src[(size_t)m * lds + n] : 0.f);
}

// reparameterize: z, w_obs; build zw f32 (112) and zw bf16 (128, padded)
__global__ void reparam_kernel(const float* __restrict__ z_mu, const float* __restrict__ z_lv,
                               const float* __restrict__ eps_z,
                               const float* __restrict__ w_mu, const float* __restrict__ w_lv,
                               const float* __restrict__ eps_w,
                               float* __restrict__ z_f, float* __restrict__ zw_f,
                               bf16_t* __restrict__ zw_bf)
{
    int idx = blockIdx.x * 256 + threadIdx.x;
    if (idx >= NB * 128) return;
    int b = idx >> 7, j = idx & 127;
    float v = 0.f;
    if (j < NZ) {
        int k = b * NZ + j;
        v = z_mu[k] + eps_z[k] * expf(0.5f * z_lv[k]);
        z_f[k] = v;
        zw_f[b * NZW + j] = v;
    } else if (j < NZW) {
        int k = b * NW + (j - NZ);
        v = w_mu[k] + eps_w[k] * expf(0.5f * w_lv[k]);
        zw_f[b * NZW + j] = v;
    }
    zw_bf[idx] = (bf16_t)((j < NZW) ? v : 0.f);
}

// ---------------------------------------------------------------------------
// Losses (block reduce + atomicAdd into accumulators)
// acc[0]=sum lp, acc[1]=sum wkl, acc[2]=sum zkl, acc[3]=sum negent, acc[4]=sum yrecon
// ---------------------------------------------------------------------------
__global__ void zinb_loss_kernel(const float* __restrict__ x,
                                 const float* __restrict__ mu_out,
                                 const float* __restrict__ gate_out,
                                 const float* __restrict__ log_theta,
                                 float* __restrict__ acc)
{
    __shared__ float red[256];
    int idx = blockIdx.x * 256 + threadIdx.x;
    float s = 0.f;
    if (idx < NB * NG) {
        int g = idx % NG;
        float xv   = x[idx];
        float th   = expf(log_theta[g]);
        float mu   = expf(mu_out[idx]);
        float nb   = logf(mu + 1e-5f) - logf(th + 1e-5f);
        float nblp = lgammaf(xv + th) - lgammaf(1.f + xv) - lgammaf(th)
                   + th * logsigf(-nb) + xv * logsigf(nb);
        float gl       = gate_out[idx];
        float log_gate = -softplusf(-gl);
        float lp_mlg   = -gl + nblp;
        s = (xv == 0.f) ? (softplusf(lp_mlg) + log_gate)
                        : (lp_mlg + log_gate);
    }
    red[threadIdx.x] = s;
    __syncthreads();
    for (int o = 128; o; o >>= 1) {
        if (threadIdx.x < o) red[threadIdx.x] += red[threadIdx.x + o];
        __syncthreads();
    }
    if (threadIdx.x == 0) atomicAdd(&acc[0], red[0]);
}

__global__ void wkl_kernel(const float* __restrict__ wm, const float* __restrict__ wl,
                           const float* __restrict__ pm, const float* __restrict__ pl,
                           float* __restrict__ acc)
{
    __shared__ float red[256];
    int idx = blockIdx.x * 256 + threadIdx.x;
    float s = 0.f;
    if (idx < NB * NW) {
        float d = wm[idx] - pm[idx];
        s = pl[idx] - wl[idx] + (expf(wl[idx]) + d * d) * expf(-pl[idx]) - 1.f;
    }
    red[threadIdx.x] = s;
    __syncthreads();
    for (int o = 128; o; o >>= 1) {
        if (threadIdx.x < o) red[threadIdx.x] += red[threadIdx.x + o];
        __syncthreads();
    }
    if (threadIdx.x == 0) atomicAdd(&acc[1], red[0]);
}

__global__ void zkl_kernel(const float* __restrict__ zm, const float* __restrict__ zl,
                           float* __restrict__ acc)
{
    __shared__ float red[256];
    int idx = blockIdx.x * 256 + threadIdx.x;
    float s = 0.f;
    if (idx < NB * NZ)
        s = -zl[idx] + expf(zl[idx]) + zm[idx] * zm[idx] - 1.f;
    red[threadIdx.x] = s;
    __syncthreads();
    for (int o = 128; o; o >>= 1) {
        if (threadIdx.x < o) red[threadIdx.x] += red[threadIdx.x + o];
        __syncthreads();
    }
    if (threadIdx.x == 0) atomicAdd(&acc[2], red[0]);
}

__global__ void yloss_kernel(const float* __restrict__ y, const float* __restrict__ zy_out,
                             float* __restrict__ acc)
{
    __shared__ float rne[256];
    __shared__ float rre[256];
    int idx = blockIdx.x * 256 + threadIdx.x;
    float sne = 0.f, sre = 0.f;
    if (idx < NB * NY) {
        float t  = zy_out[idx];
        float yp = 1.f / (1.f + expf(-t));
        yp = fminf(fmaxf(yp, 1e-12f), 1.f - 1e-12f);
        float lyp = logf(yp), l1m = log1pf(-yp);
        sne = lyp * yp + l1m * (1.f - yp);
        float yv = y[idx];
        sre = yv * lyp + (1.f - yv) * l1m;
    }
    rne[threadIdx.x] = sne;
    rre[threadIdx.x] = sre;
    __syncthreads();
    for (int o = 128; o; o >>= 1) {
        if (threadIdx.x < o) { rne[threadIdx.x] += rne[threadIdx.x + o];
                               rre[threadIdx.x] += rre[threadIdx.x + o]; }
        __syncthreads();
    }
    if (threadIdx.x == 0) { atomicAdd(&acc[3], rne[0]); atomicAdd(&acc[4], rre[0]); }
}

__global__ void zero_acc_kernel(float* acc) {
    if (threadIdx.x < 8) acc[threadIdx.x] = 0.f;
}

__global__ void finalize_kernel(const float* __restrict__ acc, float* __restrict__ out)
{
    const float inv_b  = 1.f / (float)NB;
    const float inv_by = 1.f / (float)(NB * NY);
    float x_recon = -(acc[0] * inv_b) * 1.0f;       // BETA0
    float w_kl    = 0.5f * acc[1] * 1.0f;           // BETA1
    float z_kl    = 0.5f * acc[2] * 0.1f;           // BETA2
    float yne     = (acc[3] * inv_by) * 1.0f;       // BETA3
    float yre     = -(acc[4] * inv_by) * 1.0f;      // BETA4
    out[0] = x_recon + w_kl + z_kl + yne;
    out[1] = x_recon;
    out[2] = w_kl;
    out[3] = z_kl;
    out[4] = yne;
    out[5] = yre;
}

// ---------------------------------------------------------------------------
extern "C" void kernel_launch(void* const* d_in, const int* in_sizes, int n_in,
                              void* d_out, int out_size, void* d_ws, size_t ws_size,
                              hipStream_t stream)
{
    (void)in_sizes; (void)n_in; (void)out_size; (void)ws_size;
    const float* x         = (const float*)d_in[0];
    const float* y         = (const float*)d_in[1];
    const float* eps_w     = (const float*)d_in[2];
    const float* eps_z     = (const float*)d_in[3];
    const float* log_theta = (const float*)d_in[86];

    char* base = (char*)d_ws;
    size_t off = 0;
    auto alloc = [&](size_t bytes) -> char* {
        char* p = base + off;
        off += (bytes + 255) & ~(size_t)255;
        return p;
    };

    // ---- workspace layout ----
    float*  acc      = (float*)alloc(8 * sizeof(float));
    bf16_t* xy_bf    = (bf16_t*)alloc((size_t)NB * 4032 * 2);
    bf16_t* y_bf     = (bf16_t*)alloc((size_t)NB * 32 * 2);
    bf16_t* ha       = (bf16_t*)alloc((size_t)NB * NH * 2);
    bf16_t* hb       = (bf16_t*)alloc((size_t)NB * NH * 2);
    float*  w_lat    = (float*)alloc((size_t)NB * NW * 4);
    float*  z_lat    = (float*)alloc((size_t)NB * NZ * 4);
    float*  wp_lat   = (float*)alloc((size_t)NB * NW * 4);
    float*  tA       = (float*)alloc((size_t)NB * NZW * 4);
    float*  tB       = (float*)alloc((size_t)NB * NZW * 4);
    float*  w_mu_f   = (float*)alloc((size_t)NB * NW * 4);
    float*  w_lv_f   = (float*)alloc((size_t)NB * NW * 4);
    float*  wp_mu_f  = (float*)alloc((size_t)NB * NW * 4);
    float*  wp_lv_f  = (float*)alloc((size_t)NB * NW * 4);
    float*  z_mu_f   = (float*)alloc((size_t)NB * NZ * 4);
    float*  z_lv_f   = (float*)alloc((size_t)NB * NZ * 4);
    float*  z_f      = (float*)alloc((size_t)NB * NZ * 4);
    float*  zw_f     = (float*)alloc((size_t)NB * NZW * 4);
    float*  xlat_f   = (float*)alloc((size_t)NB * NZW * 4);
    bf16_t* zw_bf    = (bf16_t*)alloc((size_t)NB * 128 * 2);
    bf16_t* xlat_bf  = (bf16_t*)alloc((size_t)NB * 128 * 2);
    bf16_t* z2_bf    = (bf16_t*)alloc((size_t)NB * 32 * 2);
    float*  mu_out   = (float*)alloc((size_t)NB * NG * 4);
    float*  gate_out = (float*)alloc((size_t)NB * NG * 4);
    float*  zy_out   = (float*)alloc((size_t)NB * NY * 4);
    // bf16 weights, TRANSPOSED [Np, Kp], zero-padded
    bf16_t* wl1 = (bf16_t*)alloc((size_t)1024 * 4032 * 2);
    bf16_t* wl2 = (bf16_t*)alloc((size_t)1024 * 1024 * 2);
    bf16_t* wl3 = (bf16_t*)alloc((size_t)128  * 1024 * 2);
    bf16_t* zl1 = (bf16_t*)alloc((size_t)1024 * 4000 * 2);
    bf16_t* zl2 = (bf16_t*)alloc((size_t)1024 * 1024 * 2);
    bf16_t* zl3 = (bf16_t*)alloc((size_t)128  * 1024 * 2);
    bf16_t* wp1 = (bf16_t*)alloc((size_t)1024 * 32   * 2);
    bf16_t* wp2 = (bf16_t*)alloc((size_t)1024 * 1024 * 2);
    bf16_t* wp3 = (bf16_t*)alloc((size_t)128  * 1024 * 2);
    bf16_t* mx1 = (bf16_t*)alloc((size_t)1024 * 128  * 2);
    bf16_t* mx2 = (bf16_t*)alloc((size_t)1024 * 1024 * 2);
    bf16_t* mx3 = (bf16_t*)alloc((size_t)4096 * 1024 * 2);
    bf16_t* gx1 = (bf16_t*)alloc((size_t)1024 * 128  * 2);
    bf16_t* gx2 = (bf16_t*)alloc((size_t)1024 * 1024 * 2);
    bf16_t* gx3 = (bf16_t*)alloc((size_t)4096 * 1024 * 2);
    bf16_t* zy3 = (bf16_t*)alloc((size_t)1024 * 32   * 2);
    bf16_t* zy4 = (bf16_t*)alloc((size_t)1024 * 1024 * 2);
    bf16_t* zy5 = (bf16_t*)alloc((size_t)128  * 1024 * 2);

    auto cvtw = [&](int widx, bf16_t* dst, int K, int N, int Kp, int Np) {
        int total = Np * Kp;
        convert_wt_kernel<<<(total + 255) / 256, 256, 0, stream>>>(
            (const float*)d_in[widx], dst, K, N, Kp, Np);
    };
    auto gemm = [&](const bf16_t* A, int lda, const bf16_t* Bt, int ldbt, int bidx,
                    float* Cf, int ldc, bf16_t* Cb, int ldcb, int N, int Kp, int act) {
        dim3 g((N + 127) / 128, NB / 128);
        gemm_bf16_kernel<<<g, 256, 0, stream>>>(A, lda, Bt, ldbt,
            (const float*)d_in[bidx], Cf, ldc, Cb, ldcb, NB, N, Kp, act);
    };
    auto dense = [&](const float* X, int ldx, int Widx, float* Yo, int ldy,
                     int N, int K, int act) {
        int total = NB * N;
        dense_f32_kernel<<<(total + 255) / 256, 256, 0, stream>>>(
            X, ldx, (const float*)d_in[Widx], (const float*)d_in[Widx + 1],
            Yo, ldy, NB, N, K, act);
    };

    zero_acc_kernel<<<1, 32, 0, stream>>>(acc);

    // weight conversions (f32 [K,N] -> bf16 [Np,Kp] transposed, zero-padded)
    cvtw(4,  wl1, 4010, 1024, 4032, 1024);
    cvtw(6,  wl2, 1024, 1024, 1024, 1024);
    cvtw(8,  wl3, 1024,   80, 1024,  128);
    cvtw(22, zl1, 4000, 1024, 4000, 1024);
    cvtw(24, zl2, 1024, 1024, 1024, 1024);
    cvtw(26, zl3, 1024,   32, 1024,  128);
    cvtw(40, wp1,   10, 1024,   32, 1024);
    cvtw(42, wp2, 1024, 1024, 1024, 1024);
    cvtw(44, wp3, 1024,   80, 1024,  128);
    cvtw(64, mx1,  112, 1024,  128, 1024);
    cvtw(66, mx2, 1024, 1024, 1024, 1024);
    cvtw(68, mx3, 1024, 4000, 1024, 4096);
    cvtw(70, gx1,  112, 1024,  128, 1024);
    cvtw(72, gx2, 1024, 1024, 1024, 1024);
    cvtw(74, gx3, 1024, 4000, 1024, 4096);
    cvtw(80, zy3,   32, 1024,   32, 1024);
    cvtw(82, zy4, 1024, 1024, 1024, 1024);
    cvtw(84, zy5, 1024,   10, 1024,  128);

    prep_xy_kernel<<<(NB * 4032 + 255) / 256, 256, 0, stream>>>(x, y, xy_bf);
    prep_y_kernel<<<(NB * 32 + 255) / 256, 256, 0, stream>>>(y, y_bf);

    // w_latent: xy -> H -> H -> softplus -> [B,80]
    gemm(xy_bf, 4032, wl1, 4032, 5, nullptr, 0, ha, 1024, 1024, 4032, 1);
    gemm(ha,    1024, wl2, 1024, 7, nullptr, 0, hb, 1024, 1024, 1024, 1);
    gemm(hb,    1024, wl3, 1024, 9, w_lat,  80, nullptr, 0,  80, 1024, 2);
    dense(w_lat, 80, 10, tA, 80, 80, 80, 1);
    dense(tA,    80, 12, tB, 80, 80, 80, 1);
    dense(tB,    80, 14, w_mu_f, 80, 80, 80, 0);
    dense(w_lat, 80, 16, tA, 80, 80, 80, 1);
    dense(tA,    80, 18, tB, 80, 80, 80, 1);
    dense(tB,    80, 20, w_lv_f, 80, 80, 80, 0);

    // z_latent: x_log -> H -> H -> softplus -> [B,32]
    gemm(xy_bf, 4032, zl1, 4000, 23, nullptr, 0, ha, 1024, 1024, 4000, 1);
    gemm(ha,    1024, zl2, 1024, 25, nullptr, 0, hb, 1024, 1024, 1024, 1);
    gemm(hb,    1024, zl3, 1024, 27, z_lat,  32, nullptr, 0,  32, 1024, 2);
    dense(z_lat, 32, 28, tA, 32, 32, 32, 1);
    dense(tA,    32, 30, tB, 32, 32, 32, 1);
    dense(tB,    32, 32, z_mu_f, 32, 32, 32, 0);
    dense(z_lat, 32, 34, tA, 32, 32, 32, 1);
    dense(tA,    32, 36, tB, 32, 32, 32, 1);
    dense(tB,    32, 38, z_lv_f, 32, 32, 32, 0);

    // w_prior: y -> H -> H -> softplus -> [B,80]
    gemm(y_bf,    32, wp1,   32, 41, nullptr, 0, ha, 1024, 1024,   32, 1);
    gemm(ha,    1024, wp2, 1024, 43, nullptr, 0, hb, 1024, 1024, 1024, 1);
    gemm(hb,    1024, wp3, 1024, 45, wp_lat, 80, nullptr, 0,  80, 1024, 2);
    dense(wp_lat, 80, 46, tA, 80, 80, 80, 1);
    dense(tA,     80, 48, tB, 80, 80, 80, 1);
    dense(tB,     80, 50, wp_mu_f, 80, 80, 80, 0);
    dense(wp_lat, 80, 52, tA, 80, 80, 80, 1);
    dense(tA,     80, 54, tB, 80, 80, 80, 1);
    dense(tB,     80, 56, wp_lv_f, 80, 80, 80, 0);

    reparam_kernel<<<(NB * 128 + 255) / 256, 256, 0, stream>>>(
        z_mu_f, z_lv_f, eps_z, w_mu_f, w_lv_f, eps_w, z_f, zw_f, zw_bf);

    // x_latent (112-wide) + softplus, then bf16 pad to 128
    dense(zw_f, 112, 58, tA, 112, 112, 112, 1);
    dense(tA,   112, 60, tB, 112, 112, 112, 1);
    dense(tB,   112, 62, xlat_f, 112, 112, 112, 2);
    cvt_act_kernel<<<(NB * 128 + 255) / 256, 256, 0, stream>>>(
        xlat_f, 112, xlat_bf, 128, 112, NB * 128);

    // mu_x chain -> mu_out (pre-exp, f32)
    gemm(xlat_bf, 128, mx1, 1024, 65, nullptr, 0, ha, 1024, 1024,  128, 1);
    gemm(ha,     1024, mx2, 1024, 67, nullptr, 0, hb, 1024, 1024, 1024, 1);
    gemm(hb,     1024, mx3, 1024, 69, mu_out, 4000, nullptr, 0, 4000, 1024, 0);
    // logits_x chain -> gate_out
    gemm(xlat_bf, 128, gx1, 1024, 71, nullptr, 0, ha, 1024, 1024,  128, 1);
    gemm(ha,     1024, gx2, 1024, 73, nullptr, 0, hb, 1024, 1024, 1024, 1);
    gemm(hb,     1024, gx3, 1024, 75, gate_out, 4000, nullptr, 0, 4000, 1024, 0);

    // z_y chain: two 32-wide dense, then WMMA 32->H->H->10
    dense(z_f, 32, 76, tA, 32, 32, 32, 1);
    dense(tA,  32, 78, tB, 32, 32, 32, 1);
    cvt_act_kernel<<<(NB * 32 + 255) / 256, 256, 0, stream>>>(
        tB, 32, z2_bf, 32, 32, NB * 32);
    gemm(z2_bf,  32, zy3,   32, 81, nullptr, 0, ha, 1024, 1024,   32, 1);
    gemm(ha,   1024, zy4, 1024, 83, nullptr, 0, hb, 1024, 1024, 1024, 1);
    gemm(hb,   1024, zy5, 1024, 85, zy_out, 10, nullptr, 0, 10, 1024, 0);

    // losses
    zinb_loss_kernel<<<(NB * NG + 255) / 256, 256, 0, stream>>>(
        x, mu_out, gate_out, log_theta, acc);
    wkl_kernel<<<(NB * NW + 255) / 256, 256, 0, stream>>>(
        w_mu_f, w_lv_f, wp_mu_f, wp_lv_f, acc);
    zkl_kernel<<<(NB * NZ + 255) / 256, 256, 0, stream>>>(z_mu_f, z_lv_f, acc);
    yloss_kernel<<<(NB * NY + 255) / 256, 256, 0, stream>>>(y, zy_out, acc);
    finalize_kernel<<<1, 1, 0, stream>>>(acc, (float*)d_out);
}
